// HDLoss_8383776162446
// MI455X (gfx1250) — compile-verified, matrix operand order checked by reading, not run on hardware
//
#include <hip/hip_runtime.h>

typedef int v8i __attribute__((ext_vector_type(8)));

#define HH 256
#define WW 256
#define NPIX 65536
#define NWORDS 2048
#define INF16 4096u
#define MAXHD 14500.0f

__device__ __forceinline__ unsigned umin32(unsigned a, unsigned b) { return a < b ? a : b; }

// spread 4 bits into 4 bytes (0/1 each)
__device__ __forceinline__ unsigned spread4(unsigned x) {
    return (x & 1u) | ((x & 2u) << 7) | ((x & 4u) << 14) | ((x & 8u) << 21);
}

// expand one 32-bit mask word into 32 bytes (v8i) of 0/1 for the IU8 WMMA A-operand.
// Exact byte->(m,k) placement is irrelevant: B is all-ones, so D row-sums count every byte.
__device__ __forceinline__ v8i expand_word(unsigned w) {
    v8i a;
#pragma unroll
    for (int g = 0; g < 8; ++g) a[g] = (int)spread4((w >> (4 * g)) & 0xFu);
    return a;
}

// Popcount of a 2048-word LDS bitset via v_wmma_i32_16x16x64_iu8.
// All 1024 threads must execute this (EXEC all-ones for WMMA).
// Accumulates 16 * popcount into *cnt (each of the 16 B-columns replicates the row sums).
__device__ __forceinline__ void wmma_popcount(const unsigned* bits, unsigned* cnt) {
    const int wave = threadIdx.x >> 5;
    const int lane = threadIdx.x & 31;
    v8i ones;
#pragma unroll
    for (int g = 0; g < 8; ++g) ones[g] = 0x01010101;
    v8i acc;
#pragma unroll
    for (int g = 0; g < 8; ++g) acc[g] = 0;
    acc = __builtin_amdgcn_wmma_i32_16x16x64_iu8(
        false, expand_word(bits[wave * 64 + lane]), false, ones, acc, false, false);
    acc = __builtin_amdgcn_wmma_i32_16x16x64_iu8(
        false, expand_word(bits[wave * 64 + 32 + lane]), false, ones, acc, false, false);
    int s = 0;
#pragma unroll
    for (int g = 0; g < 8; ++g) s += acc[g];
#pragma unroll
    for (int off = 16; off > 0; off >>= 1) s += __shfl_down(s, off, 32);
    if (lane == 0) atomicAdd(cnt, (unsigned)s);
}

// Build the per-row 1D nearest-border-distance field d1 (u16, INF16 sentinel) from a
// border bitset, 32-row LDS bands, exact min-plus doubling (s = 1,2,...,128).
__device__ __forceinline__ void build_d1(const unsigned* bits, unsigned short* band,
                                         unsigned short* __restrict__ out) {
    const int tid = threadIdx.x;
    for (int b0 = 0; b0 < HH; b0 += 32) {
        for (int c = tid; c < 32 * WW; c += 1024) {
            int r = b0 + (c >> 8), j = c & 255;
            unsigned wrd = bits[(r << 3) | (j >> 5)];
            band[c] = (unsigned short)(((wrd >> (j & 31)) & 1u) ? 0u : INF16);
        }
        __syncthreads();
#pragma unroll
        for (int s = 1; s < 256; s <<= 1) {
            unsigned v[8];
#pragma unroll
            for (int k = 0; k < 8; ++k) {
                int c = tid + k * 1024;
                int j = c & 255;
                unsigned cur = band[c];
                if (j >= s)       cur = umin32(cur, (unsigned)band[c - s] + (unsigned)s);
                if (j + s < 256)  cur = umin32(cur, (unsigned)band[c + s] + (unsigned)s);
                v[k] = cur;
            }
#pragma unroll
            for (int k = 0; k < 8; ++k) band[tid + k * 1024] = (unsigned short)v[k];
            __syncthreads();
        }
        for (int c = tid; c < 32 * WW; c += 1024) out[b0 * WW + c] = band[c];
        __syncthreads();
    }
}

// Vertical parabolic pass at one pixel: min over rows of d1[r][j]^2 + (i-r)^2.
__device__ __forceinline__ unsigned pass2_min(const unsigned short* __restrict__ d1, int p) {
    int i = p >> 8, j = p & 255;
    unsigned best = 0xFFFFFFFFu;
#pragma unroll 4
    for (int r = 0; r < HH; ++r) {
        unsigned d = (unsigned)d1[(r << 8) | j];
        int di = i - r;
        best = umin32(best, d * d + (unsigned)(di * di));
    }
    return best;
}

__global__ __launch_bounds__(1024) void hd95_kernel(const float* __restrict__ preds,
                                                    const float* __restrict__ targets,
                                                    unsigned short* __restrict__ d1ws,
                                                    float* __restrict__ res) {
    __shared__ unsigned sBitsP[NWORDS];   // pred mask -> pred border bits
    __shared__ unsigned sBitsT[NWORDS];   // target mask -> target border bits
    __shared__ unsigned short sBand[32 * WW];
    __shared__ unsigned sHist[512];
    __shared__ unsigned sFineA[256];
    __shared__ unsigned sFineB[256];
    __shared__ unsigned sCnt[2];
    __shared__ int sAny[2];
    __shared__ int sSel[2];

    const int sample = blockIdx.x;
    const int tid = threadIdx.x;
    const float* P = preds + (size_t)sample * NPIX;
    const float* T = targets + (size_t)sample * NPIX;
    unsigned short* d1T = d1ws + (size_t)sample * (2 * NPIX); // dist-to-target-border 1D field
    unsigned short* d1P = d1T + NPIX;                         // dist-to-pred-border 1D field

    if (tid < 2) { sCnt[tid] = 0; sAny[tid] = 0; }

    // ---- P0: threshold -> mask bitsets via wave ballot (coalesced loads) ----
    unsigned accAnyP = 0, accAnyT = 0;
    for (int base = 0; base < NPIX; base += 1024) {
        int p = base + tid;
        if (base + 8192 < NPIX) {
            __builtin_prefetch(&P[base + 8192 + tid], 0, 1);
            __builtin_prefetch(&T[base + 8192 + tid], 0, 1);
        }
        bool mp = P[p] > 0.5f;
        bool mt = T[p] > 0.5f;
        unsigned bp = (unsigned)__ballot(mp);
        unsigned bt = (unsigned)__ballot(mt);
        if ((tid & 31) == 0) {
            sBitsP[p >> 5] = bp;
            sBitsT[p >> 5] = bt;
            accAnyP |= bp;
            accAnyT |= bt;
        }
    }
    if ((tid & 31) == 0) {
        if (accAnyP) atomicOr(&sAny[0], 1);
        if (accAnyT) atomicOr(&sAny[1], 1);
    }
    __syncthreads();

    // ---- P1: border = mask & ~erode(mask), cross element, border_value=0 ----
    unsigned bw[4];
#pragma unroll
    for (int k = 0; k < 2; ++k) {
        int w = tid + k * 1024;
        int r = w >> 3, s = w & 7;
        {
            unsigned c  = sBitsP[w];
            unsigned up = (r > 0)   ? sBitsP[w - 8] : 0u;
            unsigned dn = (r < 255) ? sBitsP[w + 8] : 0u;
            unsigned lf = (c << 1) | ((s > 0) ? (sBitsP[w - 1] >> 31) : 0u);
            unsigned rt = (c >> 1) | ((s < 7) ? (sBitsP[w + 1] << 31) : 0u);
            bw[k * 2 + 0] = c & ~(c & up & dn & lf & rt);
        }
        {
            unsigned c  = sBitsT[w];
            unsigned up = (r > 0)   ? sBitsT[w - 8] : 0u;
            unsigned dn = (r < 255) ? sBitsT[w + 8] : 0u;
            unsigned lf = (c << 1) | ((s > 0) ? (sBitsT[w - 1] >> 31) : 0u);
            unsigned rt = (c >> 1) | ((s < 7) ? (sBitsT[w + 1] << 31) : 0u);
            bw[k * 2 + 1] = c & ~(c & up & dn & lf & rt);
        }
    }
    __syncthreads();
    sBitsP[tid] = bw[0];        sBitsT[tid] = bw[1];
    sBitsP[tid + 1024] = bw[2]; sBitsT[tid + 1024] = bw[3];
    __syncthreads();

    // ---- P2: n = |pred border| + |target border| via IU8 WMMA popcount ----
    wmma_popcount(sBitsP, &sCnt[0]);
    wmma_popcount(sBitsT, &sCnt[1]);
    __syncthreads();
    const unsigned n = (sCnt[0] >> 4) + (sCnt[1] >> 4);

    // ---- P3: 1D distance fields (u16, INF sentinel) into global workspace ----
    build_d1(sBitsT, sBand, d1T);
    build_d1(sBitsP, sBand, d1P);

    // ---- P4: coarse histogram of squared surface distances (bins of 256) ----
    for (int i = tid; i < 512; i += 1024) sHist[i] = 0;
    __syncthreads();
    for (int base = 0; base < NPIX; base += 1024) {
        int p = base + tid;
        unsigned wP = sBitsP[p >> 5], wT = sBitsT[p >> 5];
        unsigned bit = 1u << (p & 31);
        if (wP & bit) { unsigned dt2 = pass2_min(d1T, p); atomicAdd(&sHist[umin32(dt2 >> 8, 511u)], 1u); }
        if (wT & bit) { unsigned dt2 = pass2_min(d1P, p); atomicAdd(&sHist[umin32(dt2 >> 8, 511u)], 1u); }
    }
    __syncthreads();

    // ---- P5: locate coarse bins holding order statistics lo and hi ----
    int rLo = 0, rHi = 0, lo = 0, hi = 0;
    float frac = 0.0f;
    if (tid < 256) { sFineA[tid] = 0; sFineB[tid] = 0; }
    if (tid == 0) {
        float pos = 0.95f * ((float)n - 1.0f);
        lo = (int)floorf(pos);
        hi = (int)ceilf(pos);
        frac = pos - (float)lo;
        int bLo = 0, bHi = 0;
        unsigned cum = 0;
        for (int b = 0; b < 512; ++b) {
            unsigned c = sHist[b];
            if (lo >= (int)cum && lo < (int)(cum + c)) { bLo = b; rLo = lo - (int)cum; }
            if (hi >= (int)cum && hi < (int)(cum + c)) { bHi = b; rHi = hi - (int)cum; }
            cum += c;
        }
        sSel[0] = bLo;
        sSel[1] = bHi;
    }
    __syncthreads();
    const int bLo = sSel[0], bHi = sSel[1];

    // ---- P6: fine histograms within the selected bins (recompute pass 2) ----
    for (int base = 0; base < NPIX; base += 1024) {
        int p = base + tid;
        unsigned wP = sBitsP[p >> 5], wT = sBitsT[p >> 5];
        unsigned bit = 1u << (p & 31);
        if (wP & bit) {
            unsigned dt2 = pass2_min(d1T, p);
            unsigned b = umin32(dt2 >> 8, 511u);
            if ((int)b == bLo) atomicAdd(&sFineA[dt2 & 255], 1u);
            if ((int)b == bHi) atomicAdd(&sFineB[dt2 & 255], 1u);
        }
        if (wT & bit) {
            unsigned dt2 = pass2_min(d1P, p);
            unsigned b = umin32(dt2 >> 8, 511u);
            if ((int)b == bLo) atomicAdd(&sFineA[dt2 & 255], 1u);
            if ((int)b == bHi) atomicAdd(&sFineB[dt2 & 255], 1u);
        }
    }
    __syncthreads();

    // ---- P7: exact order statistics -> interpolated HD95 ----
    if (tid == 0) {
        unsigned v2lo = 0, v2hi = 0, cum = 0;
        for (int idx = 0; idx < 256; ++idx) {
            unsigned c = sFineA[idx];
            if (rLo >= (int)cum && rLo < (int)(cum + c)) v2lo = ((unsigned)bLo << 8) | (unsigned)idx;
            cum += c;
        }
        cum = 0;
        for (int idx = 0; idx < 256; ++idx) {
            unsigned c = sFineB[idx];
            if (rHi >= (int)cum && rHi < (int)(cum + c)) v2hi = ((unsigned)bHi << 8) | (unsigned)idx;
            cum += c;
        }
        float hd = sqrtf((float)v2lo) * (1.0f - frac) + sqrtf((float)v2hi) * frac;
        bool valid = (sAny[0] != 0) && (sAny[1] != 0) && (n > 0);
        res[sample] = valid ? (hd * (1.0f / MAXHD)) : 1.0f;
    }
}

// Deterministic batch mean (single wave32).
__global__ void hd95_reduce_kernel(const float* __restrict__ res, float* __restrict__ out) {
    float v = res[threadIdx.x];
#pragma unroll
    for (int off = 16; off > 0; off >>= 1) v += __shfl_down(v, off, 32);
    if (threadIdx.x == 0) out[0] = v * (1.0f / 32.0f);
}

extern "C" void kernel_launch(void* const* d_in, const int* in_sizes, int n_in,
                              void* d_out, int out_size, void* d_ws, size_t ws_size,
                              hipStream_t stream) {
    (void)in_sizes; (void)n_in; (void)out_size; (void)ws_size;
    const float* preds   = (const float*)d_in[0];
    const float* targets = (const float*)d_in[1];
    // ws layout: [0, 32 * 2 * 65536 * 2B) = 8 MiB of u16 d1 fields, then 32 floats of per-sample results
    unsigned short* d1ws = (unsigned short*)d_ws;
    float* res = (float*)((char*)d_ws + (size_t)32 * 2 * 65536 * 2);
    hd95_kernel<<<32, 1024, 0, stream>>>(preds, targets, d1ws, res);
    hd95_reduce_kernel<<<1, 32, 0, stream>>>(res, (float*)d_out);
}
